// SpikingMNIST_15522011808479
// MI455X (gfx1250) — matrix operand order repaired and checked by reading, not compiled, not run on hardware
//
#include <hip/hip_runtime.h>

typedef __attribute__((ext_vector_type(16))) _Float16 v16h;
typedef __attribute__((ext_vector_type(8)))  _Float16 v8h;
typedef __attribute__((ext_vector_type(8)))  float    v8f;

#define BETA   0.95f
#define THRESH 1.0f
#define NSTEPS 25
#define BATCH  16384
#define KDIM   784
#define HID    128
#define NOUT   10

// ---------------------------------------------------------------------------
// Prologue: convert W1 [128,784] fp32 -> f16 copy in workspace (200 KB, L2-hot)
// ---------------------------------------------------------------------------
__global__ void w1_to_f16(const float* __restrict__ w, _Float16* __restrict__ wh, int n)
{
    int i = blockIdx.x * blockDim.x + threadIdx.x;
    if (i < n) wh[i] = (_Float16)w[i];
}

// ---------------------------------------------------------------------------
// Fused SNN kernel: fc1 (WMMA f16) + 25 LIF steps with fc2 (WMMA f16)
// Block = 128 threads = 4 wave32; each wave owns 16 batch rows.
// LIF identity used: reset_t = H(mem_{t-1}-1) = spike_{t-1}  (subtract reset)
// ---------------------------------------------------------------------------
__global__ __launch_bounds__(128) void snn_fused(
    const float*    __restrict__ x,     // [16384, 784]
    const _Float16* __restrict__ W1h,   // [128, 784] f16
    const float*    __restrict__ W2,    // [10, 128]
    float*          __restrict__ out)   // [25, 16384, 10]
{
    __shared__ float lds_cur1[4][16][HID];   // 32 KB: per-wave cur1 tile [m][j]

    const int wave = threadIdx.x >> 5;
    const int lane = threadIdx.x & 31;
    const int g    = lane >> 4;              // half-wave group (ISA WMMA layouts)
    const int ln   = lane & 15;
    const int rowBase = blockIdx.x * 64 + wave * 16;

    // ======================= Phase 1: cur1 = x @ W1^T =======================
    v8f acc[8] = {};                         // 8 N-tiles x (16x16 f32 C/D)

    const float* xrow = x + (size_t)(rowBase + ln) * KDIM;

    for (int c = 0; c < 24; ++c) {           // 24 full K-chunks of 32
        const int kk = c * 32;
        // A fragment (16x32, M=ln): h=0..7 -> k=kk+g*8+h ; h=8..15 -> k=kk+16+g*8+(h-8)
        const float4* p0 = (const float4*)(xrow + kk + g * 8);
        const float4* p1 = (const float4*)(xrow + kk + 16 + g * 8);
        float4 a0 = p0[0], a1 = p0[1], a2 = p1[0], a3 = p1[1];
        v16h A;
        A[0]=(_Float16)a0.x; A[1]=(_Float16)a0.y; A[2]=(_Float16)a0.z; A[3]=(_Float16)a0.w;
        A[4]=(_Float16)a1.x; A[5]=(_Float16)a1.y; A[6]=(_Float16)a1.z; A[7]=(_Float16)a1.w;
        A[8]=(_Float16)a2.x; A[9]=(_Float16)a2.y; A[10]=(_Float16)a2.z; A[11]=(_Float16)a2.w;
        A[12]=(_Float16)a3.x; A[13]=(_Float16)a3.y; A[14]=(_Float16)a3.z; A[15]=(_Float16)a3.w;
        #pragma unroll
        for (int nt = 0; nt < 8; ++nt) {
            // B fragment (32x16): col n=ln -> feature j=nt*16+ln ; h=0..7 -> k=kk+g*8+h ; h=8..15 -> k=kk+16+g*8+(h-8)
            const v8h* bp = (const v8h*)(W1h + (size_t)(nt * 16 + ln) * KDIM + kk + g * 8);
            v8h b0 = bp[0], b1 = bp[2];
            v16h B;
            #pragma unroll
            for (int i = 0; i < 8; ++i) { B[i] = b0[i]; B[8 + i] = b1[i]; }
            acc[nt] = __builtin_amdgcn_wmma_f32_16x16x32_f16(
                false, A, false, B, (short)0, acc[nt], false, false);
        }
    }
    {   // tail chunk: kk=768, valid k in [768,784); zero-pad k>=784
        const int kk = 768;
        const float4* p0 = (const float4*)(xrow + kk + g * 8);
        float4 a0 = p0[0], a1 = p0[1];
        v16h A = {};                          // halves 8..15 (k>=784) stay 0
        A[0]=(_Float16)a0.x; A[1]=(_Float16)a0.y; A[2]=(_Float16)a0.z; A[3]=(_Float16)a0.w;
        A[4]=(_Float16)a1.x; A[5]=(_Float16)a1.y; A[6]=(_Float16)a1.z; A[7]=(_Float16)a1.w;
        #pragma unroll
        for (int nt = 0; nt < 8; ++nt) {
            v16h B = {};                      // halves 8..15 cover k>=784 -> zero
            {
                const v8h* bp = (const v8h*)(W1h + (size_t)(nt * 16 + ln) * KDIM + kk + g * 8);
                v8h b0 = bp[0];
                #pragma unroll
                for (int i = 0; i < 8; ++i) { B[i] = b0[i]; }
            }
            acc[nt] = __builtin_amdgcn_wmma_f32_16x16x32_f16(
                false, A, false, B, (short)0, acc[nt], false, false);
        }
    }

    // ------- transpose cur1 from C/D layout into LDS [m][j] (once) -------
    #pragma unroll
    for (int nt = 0; nt < 8; ++nt)
        #pragma unroll
        for (int r = 0; r < 8; ++r)
            lds_cur1[wave][r + 8 * g][nt * 16 + ln] = acc[nt][r];
    __syncthreads();

    // ===================== Phase 2: LIF dynamics + fc2 ======================
    // Preload W2^T fragments f16 (B of 16x16x32): col o=ln (o<10 valid), k = c2*32+g*8+(h<8 ? h : h+8)
    v16h b2[4];
    #pragma unroll
    for (int c2 = 0; c2 < 4; ++c2) {
        #pragma unroll
        for (int h = 0; h < 16; ++h) {
            int k = c2 * 32 + g * 8 + (h < 8 ? h : h + 8);
            float w = (ln < NOUT) ? W2[ln * HID + k] : 0.0f;
            b2[c2][h] = (_Float16)w;
        }
    }

    float mem1[64];                           // row m=ln, A-fragment feature order
    #pragma unroll
    for (int i = 0; i < 64; ++i) mem1[i] = 0.0f;
    float mem2[8];                            // C/D layout: (m=r+8g, o=ln)
    float pspk2[8];                           // previous layer-2 spikes (= reset2)
    #pragma unroll
    for (int r = 0; r < 8; ++r) { mem2[r] = 0.0f; pspk2[r] = 0.0f; }

    // afrag doubles as "previous layer-1 spikes" (= reset1); spk_{-1} = 0
    v16h afrag[4] = {};

    const float* curRow = &lds_cur1[wave][ln][0];

    #pragma unroll 1
    for (int t = 0; t < NSTEPS; ++t) {
        // ---- layer-1 LIF update; prev f16 spike is the reset; new spike in f16 ----
        #pragma unroll
        for (int c2 = 0; c2 < 4; ++c2) {
            const float4* q0 = (const float4*)(curRow + c2 * 32 + g * 8);
            const float4* q1 = (const float4*)(curRow + c2 * 32 + 16 + g * 8);
            float4 u0 = q0[0], u1 = q0[1], u2 = q1[0], u3 = q1[1];
            float cv[16] = { u0.x,u0.y,u0.z,u0.w, u1.x,u1.y,u1.z,u1.w,
                             u2.x,u2.y,u2.z,u2.w, u3.x,u3.y,u3.z,u3.w };
            #pragma unroll
            for (int h = 0; h < 16; ++h) {
                int i = c2 * 16 + h;
                float m1 = mem1[i];
                // reset = THRESH * spk_prev ; spk_prev held exactly {0,1} in f16
                m1 = BETA * m1 + cv[h] - (float)afrag[c2][h];
                mem1[i] = m1;
                afrag[c2][h] = (m1 > THRESH) ? (_Float16)1.0f : (_Float16)0.0f;
            }
        }
        // ---- fc2: cur2 = spk1 @ W2^T via 4 WMMAs over K=128 ----
        v8f d2 = {};
        #pragma unroll
        for (int c2 = 0; c2 < 4; ++c2)
            d2 = __builtin_amdgcn_wmma_f32_16x16x32_f16(
                false, afrag[c2], false, b2[c2], (short)0, d2, false, false);
        // ---- layer-2 LIF update + spike output (prev spike = reset) ----
        #pragma unroll
        for (int r = 0; r < 8; ++r) {
            float m2 = mem2[r];
            m2 = BETA * m2 + d2[r] - pspk2[r];
            mem2[r] = m2;
            float spk = (m2 > THRESH) ? 1.0f : 0.0f;
            pspk2[r] = spk;
            if (ln < NOUT) {
                out[((size_t)t * BATCH + (size_t)(rowBase + r + 8 * g)) * NOUT + ln] = spk;
            }
        }
    }
}

// ---------------------------------------------------------------------------
extern "C" void kernel_launch(void* const* d_in, const int* in_sizes, int n_in,
                              void* d_out, int out_size, void* d_ws, size_t ws_size,
                              hipStream_t stream)
{
    const float* x  = (const float*)d_in[0];   // [16384,1,28,28]
    const float* W1 = (const float*)d_in[1];   // [128,784]
    const float* W2 = (const float*)d_in[2];   // [10,128]
    float* out = (float*)d_out;                // [25,16384,10]
    _Float16* W1h = (_Float16*)d_ws;           // 200,704 bytes used

    const int n1 = HID * KDIM;
    w1_to_f16<<<(n1 + 255) / 256, 256, 0, stream>>>(W1, W1h, n1);
    snn_fused<<<BATCH / 64, 128, 0, stream>>>(x, W1h, W2, out);
}
